// LowRankLinear_15221364097486
// MI455X (gfx1250) — matrix operand order repaired and checked by reading, not compiled, not run on hardware
//
#include <hip/hip_runtime.h>
#include <hip/hip_bf16.h>

typedef __attribute__((ext_vector_type(16))) __bf16 v16bf;
typedef __attribute__((ext_vector_type(8)))  __bf16 v8bf;
typedef __attribute__((ext_vector_type(8)))  float  v8f;
typedef __attribute__((ext_vector_type(4)))  float  v4f;

#define RANK  64
#define KDIM  4096
#define NDIM  4096
#define MTOT  8192        // 4 * 2048 rows
#define MTILE 64          // rows per workgroup
#define WAVES 4           // 4 wave32 -> 128 threads

// ---- helper: convert V [64,4096] and U [4096,64] f32 -> bf16 in workspace ----
__global__ __launch_bounds__(256) void cvt_bf16_kernel(const float* __restrict__ V,
                                                       const float* __restrict__ U,
                                                       __bf16* __restrict__ Vb,
                                                       __bf16* __restrict__ Ub) {
    int i = blockIdx.x * 256 + threadIdx.x;       // both arrays: 262144 elements
    Vb[i] = (__bf16)V[i];
    Ub[i] = (__bf16)U[i];
}

// load 16 contiguous bf16 (32B) -> one B/A half-K fragment
__device__ __forceinline__ v16bf load_b16(const __bf16* p) {
    v16bf b;
    *((v8bf*)&b)     = *(const v8bf*)p;
    *((v8bf*)&b + 1) = *(const v8bf*)(p + 8);
    return b;
}

// A fragment from f32 x row: contiguous K order (k = kh*16 + s), streamed non-temporally
__device__ __forceinline__ v16bf load_a_frag(const float* xrow, int k, int kh) {
    v4f x0 = __builtin_nontemporal_load((const v4f*)(xrow + k + kh * 16));
    v4f x1 = __builtin_nontemporal_load((const v4f*)(xrow + k + kh * 16 + 4));
    v4f x2 = __builtin_nontemporal_load((const v4f*)(xrow + k + kh * 16 + 8));
    v4f x3 = __builtin_nontemporal_load((const v4f*)(xrow + k + kh * 16 + 12));
    v16bf a;
    #pragma unroll
    for (int i = 0; i < 4; ++i) {
        a[i]      = (__bf16)x0[i];
        a[4 + i]  = (__bf16)x1[i];
        a[8 + i]  = (__bf16)x2[i];
        a[12 + i] = (__bf16)x3[i];
    }
    return a;
}

#define WMMA_BF16(A, B, C) \
    __builtin_amdgcn_wmma_f32_16x16x32_bf16(false, (A), false, (B), (short)0, (C), false, false)

// ---- fused low-rank linear: out = (x @ V^T) @ U^T -------------------------
__global__ __launch_bounds__(WAVES * 32)
void lowrank_fused_kernel(const float*  __restrict__ x,
                          const __bf16* __restrict__ Vb,
                          const __bf16* __restrict__ Ub,
                          float*        __restrict__ out) {
    constexpr int HP = RANK + 8;                  // pad rows (144B, 16B aligned)
    __shared__ __attribute__((aligned(16))) __bf16 hsh[MTILE][HP];

    const int lane = threadIdx.x & 31;
    const int wave = threadIdx.x >> 5;
    const int m    = lane & 15;                   // A row / B column within tile
    const int kh   = lane >> 4;                   // K-half selector
    const int rowg = blockIdx.x * MTILE + wave * 16 + m;

    // ================= GEMM1: h[16 x 64] = x[16 x 4096] @ V^T ==============
    v8f acc0 = {}, acc1 = {}, acc2 = {}, acc3 = {};
    const float*  xrow  = x + (size_t)rowg * KDIM;
    const __bf16* vbase = Vb + (size_t)m * KDIM + kh * 16;   // column m, K-half kh

    // 2-stage software pipeline: buffer0 = even chunk, buffer1 = odd chunk
    v16bf a0, a1, b00, b01, b02, b03, b10, b11, b12, b13;
    a0  = load_a_frag(xrow, 0, kh);
    b00 = load_b16(vbase);
    b01 = load_b16(vbase + 16 * KDIM);
    b02 = load_b16(vbase + 32 * KDIM);
    b03 = load_b16(vbase + 48 * KDIM);

    for (int k = 0; k < KDIM; k += 64) {
        // issue odd-chunk loads while even-chunk WMMAs run
        a1  = load_a_frag(xrow, k + 32, kh);
        b10 = load_b16(vbase + k + 32);
        b11 = load_b16(vbase + 16 * KDIM + k + 32);
        b12 = load_b16(vbase + 32 * KDIM + k + 32);
        b13 = load_b16(vbase + 48 * KDIM + k + 32);
        __builtin_prefetch(xrow + k + 1024, 0, 3);   // WGP-scope stream-ahead

        acc0 = WMMA_BF16(a0, b00, acc0);
        acc1 = WMMA_BF16(a0, b01, acc1);
        acc2 = WMMA_BF16(a0, b02, acc2);
        acc3 = WMMA_BF16(a0, b03, acc3);

        if (k + 64 < KDIM) {                          // uniform branch: EXEC intact
            a0  = load_a_frag(xrow, k + 64, kh);
            b00 = load_b16(vbase + k + 64);
            b01 = load_b16(vbase + 16 * KDIM + k + 64);
            b02 = load_b16(vbase + 32 * KDIM + k + 64);
            b03 = load_b16(vbase + 48 * KDIM + k + 64);
        }

        acc0 = WMMA_BF16(a1, b10, acc0);
        acc1 = WMMA_BF16(a1, b11, acc1);
        acc2 = WMMA_BF16(a1, b12, acc2);
        acc3 = WMMA_BF16(a1, b13, acc3);
    }

    // ====== re-layout h: C-fragments -> LDS (bf16) -> A-fragments ==========
    // C layout: VGPR j, lane l -> row j + 8*(l>>4), col l&15
    const int mrow = wave * 16 + 8 * kh;
    #pragma unroll
    for (int j = 0; j < 8; ++j) {
        hsh[mrow + j][ 0 + m] = (__bf16)acc0[j];
        hsh[mrow + j][16 + m] = (__bf16)acc1[j];
        hsh[mrow + j][32 + m] = (__bf16)acc2[j];
        hsh[mrow + j][48 + m] = (__bf16)acc3[j];
    }
    // each wave reads only its own band; per-wave LDS ordering is sufficient

    v16bf ha0, ha1;
    const __bf16* hrow = &hsh[wave * 16 + m][0];
    *((v8bf*)&ha0)     = *(const v8bf*)(hrow + kh * 16);       // r = kh*16..+7
    *((v8bf*)&ha0 + 1) = *(const v8bf*)(hrow + kh * 16 + 8);   // r = kh*16+8..+15
    *((v8bf*)&ha1)     = *(const v8bf*)(hrow + 32 + kh * 16);
    *((v8bf*)&ha1 + 1) = *(const v8bf*)(hrow + 32 + kh * 16 + 8);

    // ================= GEMM2: out[16 x 4096] = h[16 x 64] @ U^T ============
    const size_t  orow0 = (size_t)(blockIdx.x * MTILE + wave * 16 + 8 * kh) * NDIM;
    const __bf16* ubase = Ub + (size_t)m * RANK + kh * 16;     // column m, K-half kh

    v16bf u00, u01, u10, u11;                     // double-buffered U fragments
    u00 = load_b16(ubase);                        // tile 0, r = kh*16..
    u01 = load_b16(ubase + 32);                   // tile 0, r = 32+kh*16..

    for (int t = 0; t < NDIM / 16; t += 2) {
        // prefetch odd tile's U while computing even tile
        u10 = load_b16(ubase + (size_t)(t + 1) * 16 * RANK);
        u11 = load_b16(ubase + (size_t)(t + 1) * 16 * RANK + 32);

        v8f c = {};
        c = WMMA_BF16(ha0, u00, c);
        c = WMMA_BF16(ha1, u01, c);
        {
            float* op = out + orow0 + (size_t)t * 16 + m;
            #pragma unroll
            for (int j = 0; j < 8; ++j)
                __builtin_nontemporal_store(c[j], op + (size_t)j * NDIM);
        }

        if (t + 2 < NDIM / 16) {                  // uniform branch
            u00 = load_b16(ubase + (size_t)(t + 2) * 16 * RANK);
            u01 = load_b16(ubase + (size_t)(t + 2) * 16 * RANK + 32);
        }

        v8f d = {};
        d = WMMA_BF16(ha0, u10, d);
        d = WMMA_BF16(ha1, u11, d);
        {
            float* op = out + orow0 + (size_t)(t + 1) * 16 + m;
            #pragma unroll
            for (int j = 0; j < 8; ++j)
                __builtin_nontemporal_store(d[j], op + (size_t)j * NDIM);
        }
    }
}

extern "C" void kernel_launch(void* const* d_in, const int* in_sizes, int n_in,
                              void* d_out, int out_size, void* d_ws, size_t ws_size,
                              hipStream_t stream) {
    const float* x = (const float*)d_in[0];   // [4, 2048, 4096]
    const float* U = (const float*)d_in[1];   // [4096, 64]
    const float* V = (const float*)d_in[2];   // [64, 4096]
    float* out = (float*)d_out;               // [4, 2048, 4096]

    __bf16* Vb = (__bf16*)d_ws;               // 262144 bf16 = 512 KB
    __bf16* Ub = Vb + (size_t)RANK * KDIM;    // 262144 bf16 = 512 KB

    cvt_bf16_kernel<<<(RANK * KDIM) / 256, 256, 0, stream>>>(V, U, Vb, Ub);
    lowrank_fused_kernel<<<MTOT / MTILE, WAVES * 32, 0, stream>>>(x, Vb, Ub, out);
}
